// BiMPM_7172595384894
// MI455X (gfx1250) — matrix operand order repaired and checked by reading, not compiled, not run on hardware
//
#include <hip/hip_runtime.h>
#include <math.h>

// ---------------- model dims ----------------
#define NB   64          // batch B
#define NS   128         // seq len S
#define NH   128         // hidden H
#define NL   20          // perspectives L
#define ND   300         // embed dim D
#define NDP  320         // D padded to mult of 32
#define N4H  512         // 4*H
#define NAG  160         // 8*L aggregated match dim (mult of 32)
#define NC   3
#define EPSF 1e-8f

typedef __attribute__((ext_vector_type(16))) __bf16 v16bf;
typedef __attribute__((ext_vector_type(8)))  float  v8f;

union BQFrag { uint4 q[2]; v16bf v; };
union BSFrag { unsigned short s[16]; v16bf v; };

__device__ __forceinline__ unsigned short f2bf(float x) {
    unsigned int u = __float_as_uint(x);
    unsigned int r = u + 0x7FFFu + ((u >> 16) & 1u);   // RNE
    return (unsigned short)(r >> 16);
}
__device__ __forceinline__ float bf2f(unsigned short h) {
    return __uint_as_float(((unsigned int)h) << 16);
}
__device__ __forceinline__ float sigf(float x) { return 1.0f / (1.0f + expf(-x)); }

// A fragment: 16x32 bf16, row-major source (lda elems). lane m=l&15, kh=l>>4.
// element e -> k = (e&7) + 8*kh + 16*(e>>3)  => two contiguous 8-half runs.
__device__ __forceinline__ v16bf load_frag_A(const unsigned short* A, int lda, int m0, int k0) {
    int lane = threadIdx.x & 31;
    const unsigned short* row = A + (size_t)(m0 + (lane & 15)) * lda + k0 + 8 * (lane >> 4);
    BQFrag f;
    f.q[0] = *reinterpret_cast<const uint4*>(row);
    f.q[1] = *reinterpret_cast<const uint4*>(row + 16);
    return f.v;
}
// B fragment: 32x16 bf16 from N-major storage Bt[N][K] (== W row-major).
// lane n=l&15, kh=l>>4; element e -> k = (e&7) + 8*kh + 16*(e>>3) => two contiguous 8-half runs.
__device__ __forceinline__ v16bf load_frag_B(const unsigned short* Bt, int ldb, int n0, int k0) {
    int lane = threadIdx.x & 31;
    const unsigned short* row = Bt + (size_t)(n0 + (lane & 15)) * ldb + k0 + 8 * (lane >> 4);
    BQFrag f;
    f.q[0] = *reinterpret_cast<const uint4*>(row);
    f.q[1] = *reinterpret_cast<const uint4*>(row + 16);
    return f.v;
}
__device__ __forceinline__ v8f wmma_bf16(v16bf a, v16bf b, v8f c) {
    return __builtin_amdgcn_wmma_f32_16x16x32_bf16(false, a, false, b, (short)0, c, false, false);
}

// ---------------- utility kernels ----------------
__global__ void conv_bf16_k(const float* __restrict__ src, unsigned short* __restrict__ dst,
                            int rows, int cin, int cpad) {
    int idx = blockIdx.x * blockDim.x + threadIdx.x;
    if (idx >= rows * cpad) return;
    int r = idx / cpad, c = idx % cpad;
    dst[idx] = (c < cin) ? f2bf(src[(size_t)r * cin + c]) : (unsigned short)0;
}

__global__ void embed_bf16_k(const int* __restrict__ tok, const float* __restrict__ emb,
                             unsigned short* __restrict__ out) {
    int idx = blockIdx.x * blockDim.x + threadIdx.x;            // (B*S) x NDP
    if (idx >= NB * NS * NDP) return;
    int row = idx / NDP, col = idx % NDP;
    int t = tok[row];
    out[idx] = (col < ND) ? f2bf(emb[(size_t)t * ND + col]) : (unsigned short)0;
}

__global__ void zero_state_k(float* cstate, float* hstate, unsigned short* hprev) {
    int idx = blockIdx.x * blockDim.x + threadIdx.x;            // 4*64*128
    if (idx >= 4 * NB * NH) return;
    cstate[idx] = 0.0f; hstate[idx] = 0.0f; hprev[idx] = 0;
}

// ---- generic bf16 WMMA GEMM, 1x4 register blocking: C = A(MxK) * Bt(NxK)^T + bias ----
// Each wave computes one 16x64 strip: one A fragment feeds 4 WMMAs per K-chunk.
__global__ void gemm_bf16_bias_k(const unsigned short* __restrict__ A, int lda,
                                 const unsigned short* __restrict__ Bt, int ldb,
                                 const float* __restrict__ bias, float* __restrict__ C,
                                 int M, int N, int K) {
    int wave = blockIdx.x * (blockDim.x >> 5) + (threadIdx.x >> 5);
    int groupsN = N >> 6;                                        // groups of 4 N-tiles
    int strips  = (M >> 4) * groupsN;
    if (wave >= strips) return;                                  // wave-uniform
    int tm = wave / groupsN, tg = wave % groupsN;
    v8f acc[4];
#pragma unroll
    for (int s = 0; s < 4; ++s) acc[s] = (v8f){0.f,0.f,0.f,0.f,0.f,0.f,0.f,0.f};
    for (int k0 = 0; k0 < K; k0 += 32) {
        v16bf a = load_frag_A(A, lda, tm * 16, k0);
#pragma unroll
        for (int s = 0; s < 4; ++s)
            acc[s] = wmma_bf16(a, load_frag_B(Bt, ldb, tg * 64 + s * 16, k0), acc[s]);
    }
    int lane = threadIdx.x & 31;
    int mh = (lane >> 4) * 8;
#pragma unroll
    for (int s = 0; s < 4; ++s) {
        int n = tg * 64 + s * 16 + (lane & 15);
        float bv = bias ? bias[n] : 0.0f;
#pragma unroll
        for (int r = 0; r < 8; ++r) {
            int m = tm * 16 + r + mh;
            C[(size_t)m * N + n] = acc[s][r] + bv;
        }
    }
}

// ---------------- LSTM step: z = X[t] + h_prev @ U^T (WMMA, 1x4 blocked) ----------------
__global__ void lstm_step_gemm_k(const unsigned short* __restrict__ hprev,
                                 const unsigned short* __restrict__ Uf,
                                 const unsigned short* __restrict__ Ub,
                                 const float* __restrict__ Xc, float* __restrict__ zbuf, int t) {
    int job = blockIdx.y;                       // 0:s1-f 1:s1-b 2:s2-f 3:s2-b
    int dir = job & 1;
    const unsigned short* A  = hprev + (size_t)job * NB * NH;
    const unsigned short* Bt = dir ? Ub : Uf;
    int tt = dir ? (NS - 1 - t) : t;
    int wave = blockIdx.x * 8 + (threadIdx.x >> 5);   // 32 strips: 4 x 8
    int tm = wave >> 3, tg = wave & 7;
    v8f acc[4];
#pragma unroll
    for (int s = 0; s < 4; ++s) acc[s] = (v8f){0.f,0.f,0.f,0.f,0.f,0.f,0.f,0.f};
#pragma unroll
    for (int kc = 0; kc < 4; ++kc) {
        v16bf a = load_frag_A(A, NH, tm * 16, kc * 32);
#pragma unroll
        for (int s = 0; s < 4; ++s)
            acc[s] = wmma_bf16(a, load_frag_B(Bt, NH, tg * 64 + s * 16, kc * 32), acc[s]);
    }
    int lane = threadIdx.x & 31;
    int mh = (lane >> 4) * 8;
    const float* X = Xc + (size_t)job * NB * NS * N4H;
    float* z = zbuf + (size_t)job * NB * N4H;
#pragma unroll
    for (int s = 0; s < 4; ++s) {
        int n = tg * 64 + s * 16 + (lane & 15);
#pragma unroll
        for (int r = 0; r < 8; ++r) {
            int m = tm * 16 + r + mh;                            // batch index
            z[(size_t)m * N4H + n] = acc[s][r] + X[((size_t)m * NS + tt) * N4H + n];
        }
    }
}

__global__ void lstm_step_gates_k(const float* __restrict__ zbuf, float* __restrict__ cstate,
                                  unsigned short* __restrict__ hprev, float* __restrict__ hstate,
                                  float* hs0, float* hs1, float* hs2, float* hs3,
                                  unsigned short* hb0, unsigned short* hb1,
                                  unsigned short* hb2, unsigned short* hb3,
                                  int t, int storeHs) {
    int idx = blockIdx.x * blockDim.x + threadIdx.x;            // 4*64*128
    if (idx >= 4 * NB * NH) return;
    int job = idx >> 13, rem = idx & 8191;
    int b = rem >> 7, hh = rem & 127;
    const float* z = zbuf + ((size_t)job * NB + b) * N4H;
    float zi = z[hh], zf = z[NH + hh], zg = z[2 * NH + hh], zo = z[3 * NH + hh];
    float c = sigf(zf) * cstate[idx] + sigf(zi) * tanhf(zg);
    float h = sigf(zo) * tanhf(c);
    cstate[idx] = c; hstate[idx] = h; hprev[idx] = f2bf(h);
    if (storeHs) {
        int dir = job & 1;
        int tt = dir ? (NS - 1 - t) : t;
        float* hs = (job == 0) ? hs0 : (job == 1) ? hs1 : (job == 2) ? hs2 : hs3;
        unsigned short* hb = (job == 0) ? hb0 : (job == 1) ? hb1 : (job == 2) ? hb2 : hb3;
        size_t off = ((size_t)b * NS + tt) * NH + hh;
        hs[off] = h; hb[off] = f2bf(h);
    }
}

// ---------------- norms ----------------
__global__ void bs_norm_k(const float* __restrict__ x, float* __restrict__ n) {
    int idx = blockIdx.x * blockDim.x + threadIdx.x;            // B*S
    if (idx >= NB * NS) return;
    const float* row = x + (size_t)idx * NH;
    float s = 0.f;
    for (int h = 0; h < NH; ++h) { float v = row[h]; s += v * v; }
    n[idx] = sqrtf(s);
}
__global__ void w_norm_k(const float* __restrict__ x, const float* __restrict__ w,
                         float* __restrict__ out) {
    int idx = blockIdx.x * blockDim.x + threadIdx.x;            // B*L*S
    if (idx >= NB * NL * NS) return;
    int s = idx % NS, t = idx / NS, l = t % NL, b = t / NL;
    const float* row = x + ((size_t)b * NS + s) * NH;
    const float* wr  = w + (size_t)l * NH;
    float acc = 0.f;
    for (int h = 0; h < NH; ++h) { float v = row[h], wv = wr[h]; acc += v * v * wv * wv; }
    out[idx] = sqrtf(acc);
}

// ---------------- full match ----------------
__global__ void full_match_k(const float* __restrict__ p1, const float* __restrict__ p2base,
                             int p2stride, const float* __restrict__ w,
                             float* __restrict__ aggr, int col) {
    int idx = blockIdx.x * blockDim.x + threadIdx.x;            // B*S*L
    if (idx >= NB * NS * NL) return;
    int l = idx % NL, t = idx / NL, s = t % NS, b = t / NS;
    const float* a = p1 + ((size_t)b * NS + s) * NH;
    const float* c = p2base + (size_t)b * p2stride;
    const float* wr = w + (size_t)l * NH;
    float dot = 0.f, na = 0.f, nb = 0.f;
    for (int h = 0; h < NH; ++h) {
        float w2 = wr[h] * wr[h];
        dot += a[h] * c[h] * w2; na += a[h] * a[h] * w2; nb += c[h] * c[h] * w2;
    }
    aggr[((size_t)b * NS + s) * NAG + col + l] =
        dot / (fmaxf(sqrtf(na), EPSF) * fmaxf(sqrtf(nb), EPSF));
}

// ---------------- maxpool match (WMMA, fused max over j, A reused across j-tiles) -------
__global__ void maxpool_match_k(const unsigned short* __restrict__ p1b,
                                const unsigned short* __restrict__ p2b,
                                const unsigned short* __restrict__ wb,
                                const float* __restrict__ n1w, const float* __restrict__ n2w,
                                float* __restrict__ aggr, int col) {
    int wave = blockIdx.x * 8 + (threadIdx.x >> 5);             // B*L*(S/16) waves
    if (wave >= NB * NL * (NS / 16)) return;
    int it = wave & 7, bl = wave >> 3, l = bl % NL, b = bl / NL;
    int lane = threadIdx.x & 31, kh = lane >> 4, mrow = lane & 15;
    // A = p1[b, it*16+m, :] * w[l, :]  (built in-register, bf16)
    v16bf afr[4];
#pragma unroll
    for (int kc = 0; kc < 4; ++kc) {
        const unsigned short* pr = p1b + ((size_t)b * NS + it * 16 + mrow) * NH + kc * 32 + 8 * kh;
        const unsigned short* wr = wb + (size_t)l * NH + kc * 32 + 8 * kh;
        BSFrag f;
#pragma unroll
        for (int e = 0; e < 8; ++e) {
            f.s[e]     = f2bf(bf2f(pr[e])      * bf2f(wr[e]));
            f.s[8 + e] = f2bf(bf2f(pr[16 + e]) * bf2f(wr[16 + e]));
        }
        afr[kc] = f.v;
    }
    float n1v[8], rmax[8];
#pragma unroll
    for (int r = 0; r < 8; ++r) {
        n1v[r] = n1w[((size_t)b * NL + l) * NS + it * 16 + r + 8 * kh];
        rmax[r] = -1e30f;
    }
    const unsigned short* Bbase = p2b + (size_t)b * NS * NH;
    for (int jt = 0; jt < NS / 16; ++jt) {
        v8f acc = {0.f,0.f,0.f,0.f,0.f,0.f,0.f,0.f};
#pragma unroll
        for (int kc = 0; kc < 4; ++kc)
            acc = wmma_bf16(afr[kc], load_frag_B(Bbase, NH, jt * 16, kc * 32), acc);
        float n2v = n2w[((size_t)b * NL + l) * NS + jt * 16 + (lane & 15)];
#pragma unroll
        for (int r = 0; r < 8; ++r) {
            float deno = n1v[r] * n2v;
            deno = (deno > EPSF) ? deno : EPSF;
            rmax[r] = fmaxf(rmax[r], acc[r] / deno);
        }
    }
#pragma unroll
    for (int r = 0; r < 8; ++r) {
        for (int off = 1; off < 16; off <<= 1)
            rmax[r] = fmaxf(rmax[r], __shfl_xor(rmax[r], off, 32));
        if ((lane & 15) == 0) {
            int i = it * 16 + r + 8 * kh;
            aggr[((size_t)b * NS + i) * NAG + col + l] = rmax[r];
        }
    }
}

// ---------------- attentive match ----------------
// alpha GEMM per s: (64x128)x(128x64); wave computes a 16x64 strip (4 accs, A reused).
__global__ void att_alpha_k(const unsigned short* __restrict__ p1b,
                            const unsigned short* __restrict__ p2b,
                            const float* __restrict__ n1, const float* __restrict__ n2,
                            float* __restrict__ alpha) {
    int wave = blockIdx.x * 8 + (threadIdx.x >> 5);             // S * 4 strips
    if (wave >= NS * 4) return;
    int s = wave >> 2, tm = wave & 3;
    const unsigned short* A  = p1b + (size_t)s * NH;            // rows stride S*H
    const unsigned short* Bt = p2b + (size_t)s * NH;
    v8f acc[4];
#pragma unroll
    for (int q = 0; q < 4; ++q) acc[q] = (v8f){0.f,0.f,0.f,0.f,0.f,0.f,0.f,0.f};
#pragma unroll
    for (int kc = 0; kc < 4; ++kc) {
        v16bf a = load_frag_A(A, NS * NH, tm * 16, kc * 32);
#pragma unroll
        for (int q = 0; q < 4; ++q)
            acc[q] = wmma_bf16(a, load_frag_B(Bt, NS * NH, q * 16, kc * 32), acc[q]);
    }
    int lane = threadIdx.x & 31;
    int mh = (lane >> 4) * 8;
#pragma unroll
    for (int q = 0; q < 4; ++q) {
        int n = q * 16 + (lane & 15);
#pragma unroll
        for (int r = 0; r < 8; ++r) {
            int m = tm * 16 + r + mh;
            float deno = n1[(size_t)m * NS + s] * n2[(size_t)n * NS + s];
            deno = (deno > EPSF) ? deno : EPSF;
            alpha[((size_t)s * NB + m) * NB + n] = acc[q][r] / deno;
        }
    }
}
__global__ void att_soft_k(const float* __restrict__ alpha, float* __restrict__ rsum,
                           int* __restrict__ amax) {
    int idx = blockIdx.x * blockDim.x + threadIdx.x;            // S*B
    if (idx >= NS * NB) return;
    const float* row = alpha + (size_t)idx * NB;
    float s = 0.f, best = -1e30f; int bi = 0;
    for (int c = 0; c < NB; ++c) {
        float v = row[c]; s += v;
        if (v > best) { best = v; bi = c; }
    }
    rsum[idx] = s; amax[idx] = bi;
}
__global__ void att_h_k(const float* __restrict__ alpha, const float* __restrict__ rsum,
                        const float* __restrict__ p2, float* __restrict__ rbuf) {
    int idx = blockIdx.x * blockDim.x + threadIdx.x;            // B*S*H
    if (idx >= NB * NS * NH) return;
    int b = idx >> 14, rem = idx & 16383, s = rem >> 7, hh = rem & 127;
    const float* arow = alpha + ((size_t)s * NB + b) * NB;
    float acc = 0.f;
    for (int c = 0; c < NB; ++c)
        acc += arow[c] * p2[((size_t)c * NS + s) * NH + hh];
    rbuf[idx] = acc / rsum[(size_t)s * NB + b];
}
__global__ void att_match_k(const float* __restrict__ rbuf, const float* __restrict__ p1,
                            const float* __restrict__ w, float* __restrict__ aggr, int col) {
    int idx = blockIdx.x * blockDim.x + threadIdx.x;            // B*S*L
    if (idx >= NB * NS * NL) return;
    int l = idx % NL, t = idx / NL, s = t % NS, b = t / NS;
    const float* a = rbuf + ((size_t)b * NS + s) * NH;
    const float* c = p1 + ((size_t)b * NS + s) * NH;
    const float* wr = w + (size_t)l * NH;
    float dot = 0.f, na = 0.f, nb = 0.f;
    for (int h = 0; h < NH; ++h) {
        float w2 = wr[h] * wr[h];
        dot += a[h] * c[h] * w2; na += a[h] * a[h] * w2; nb += c[h] * c[h] * w2;
    }
    aggr[((size_t)b * NS + s) * NAG + col + l] =
        dot / (fmaxf(sqrtf(na), EPSF) * fmaxf(sqrtf(nb), EPSF));
}
__global__ void att_max_k(const int* __restrict__ amax, const float* __restrict__ p2,
                          const float* __restrict__ p1, const float* __restrict__ w,
                          float* __restrict__ aggr, int col) {
    int idx = blockIdx.x * blockDim.x + threadIdx.x;            // B*S*L
    if (idx >= NB * NS * NL) return;
    int l = idx % NL, t = idx / NL, s = t % NS, b = t / NS;
    int j = amax[(size_t)s * NB + b];
    const float* a = p2 + ((size_t)j * NS + s) * NH;
    const float* c = p1 + ((size_t)b * NS + s) * NH;
    const float* wr = w + (size_t)l * NH;
    float dot = 0.f, na = 0.f, nb = 0.f;
    for (int h = 0; h < NH; ++h) {
        float w2 = wr[h] * wr[h];
        dot += a[h] * c[h] * w2; na += a[h] * a[h] * w2; nb += c[h] * c[h] * w2;
    }
    aggr[((size_t)b * NS + s) * NAG + col + l] =
        dot / (fmaxf(sqrtf(na), EPSF) * fmaxf(sqrtf(nb), EPSF));
}

// ---------------- final FC ----------------
__global__ void fc1_k(const float* __restrict__ hstate, const float* __restrict__ f1W,
                      const float* __restrict__ f1b, float* __restrict__ t1) {
    int idx = blockIdx.x * blockDim.x + threadIdx.x;            // B*256
    if (idx >= NB * 2 * NH) return;
    int b = idx >> 8, n = idx & 255;
    float acc = f1b[n];
    for (int k = 0; k < 4 * NH; ++k) {
        int job = k >> 7, hh = k & 127;
        acc += hstate[((size_t)job * NB + b) * NH + hh] * f1W[(size_t)n * 4 * NH + k];
    }
    t1[idx] = tanhf(acc);
}
__global__ void fc2_k(const float* __restrict__ t1, const float* __restrict__ f2W,
                      const float* __restrict__ f2b, float* __restrict__ out) {
    int idx = blockIdx.x * blockDim.x + threadIdx.x;            // B*C
    if (idx >= NB * NC) return;
    int b = idx / NC, c = idx % NC;
    float acc = f2b[c];
    for (int k = 0; k < 2 * NH; ++k)
        acc += t1[(size_t)b * 2 * NH + k] * f2W[(size_t)c * 2 * NH + k];
    out[idx] = acc;
}

// ---------------- host orchestration ----------------
extern "C" void kernel_launch(void* const* d_in, const int* in_sizes, int n_in,
                              void* d_out, int out_size, void* d_ws, size_t ws_size,
                              hipStream_t stream) {
    (void)in_sizes; (void)n_in; (void)out_size; (void)ws_size;
    const int*   p_tok = (const int*)d_in[0];
    const int*   h_tok = (const int*)d_in[1];
    const float* emb  = (const float*)d_in[2];
    const float* cWf = (const float*)d_in[3],  *cUf = (const float*)d_in[4],  *cbf = (const float*)d_in[5];
    const float* cWb = (const float*)d_in[6],  *cUb = (const float*)d_in[7],  *cbb = (const float*)d_in[8];
    const float* aWf = (const float*)d_in[9],  *aUf = (const float*)d_in[10], *abf = (const float*)d_in[11];
    const float* aWb = (const float*)d_in[12], *aUb = (const float*)d_in[13], *abb = (const float*)d_in[14];
    const float* f1W = (const float*)d_in[15], *f1b = (const float*)d_in[16];
    const float* f2W = (const float*)d_in[17], *f2b = (const float*)d_in[18];
    const float* w1 = (const float*)d_in[19], *w2 = (const float*)d_in[20];
    const float* w3 = (const float*)d_in[21], *w4 = (const float*)d_in[22];
    const float* w5 = (const float*)d_in[23], *w6 = (const float*)d_in[24];
    const float* w7 = (const float*)d_in[25], *w8 = (const float*)d_in[26];

    size_t off = 0;
    auto alloc = [&](size_t bytes) -> void* {
        void* p = (char*)d_ws + off;
        off += (bytes + 255) & ~(size_t)255;
        return p;
    };
    unsigned short* x1bf = (unsigned short*)alloc((size_t)NB*NS*NDP*2);
    unsigned short* x2bf = (unsigned short*)alloc((size_t)NB*NS*NDP*2);
    unsigned short* cWfb = (unsigned short*)alloc((size_t)N4H*NDP*2);
    unsigned short* cWbb = (unsigned short*)alloc((size_t)N4H*NDP*2);
    unsigned short* cUfb = (unsigned short*)alloc((size_t)N4H*NH*2);
    unsigned short* cUbb = (unsigned short*)alloc((size_t)N4H*NH*2);
    unsigned short* aWfb = (unsigned short*)alloc((size_t)N4H*NAG*2);
    unsigned short* aWbb = (unsigned short*)alloc((size_t)N4H*NAG*2);
    unsigned short* aUfb = (unsigned short*)alloc((size_t)N4H*NH*2);
    unsigned short* aUbb = (unsigned short*)alloc((size_t)N4H*NH*2);
    unsigned short* w3b  = (unsigned short*)alloc((size_t)NL*NH*2);
    unsigned short* w4b  = (unsigned short*)alloc((size_t)NL*NH*2);
    float* Xc    = (float*)alloc((size_t)4*NB*NS*N4H*4);        // reused for agg phase
    float* zbuf  = (float*)alloc((size_t)4*NB*N4H*4);
    float* cst   = (float*)alloc((size_t)4*NB*NH*4);
    float* hst   = (float*)alloc((size_t)4*NB*NH*4);
    unsigned short* hprev = (unsigned short*)alloc((size_t)4*NB*NH*2);
    float* c1f = (float*)alloc((size_t)NB*NS*NH*4);
    float* c1b = (float*)alloc((size_t)NB*NS*NH*4);
    float* c2f = (float*)alloc((size_t)NB*NS*NH*4);
    float* c2b = (float*)alloc((size_t)NB*NS*NH*4);
    unsigned short* c1fb = (unsigned short*)alloc((size_t)NB*NS*NH*2);
    unsigned short* c1bb = (unsigned short*)alloc((size_t)NB*NS*NH*2);
    unsigned short* c2fb = (unsigned short*)alloc((size_t)NB*NS*NH*2);
    unsigned short* c2bb = (unsigned short*)alloc((size_t)NB*NS*NH*2);
    float* wn1f3 = (float*)alloc((size_t)NB*NL*NS*4);
    float* wn2f3 = (float*)alloc((size_t)NB*NL*NS*4);
    float* wn1b4 = (float*)alloc((size_t)NB*NL*NS*4);
    float* wn2b4 = (float*)alloc((size_t)NB*NL*NS*4);
    float* nb1f = (float*)alloc((size_t)NB*NS*4);
    float* nb1b = (float*)alloc((size_t)NB*NS*4);
    float* nb2f = (float*)alloc((size_t)NB*NS*4);
    float* nb2b = (float*)alloc((size_t)NB*NS*4);
    float* aggr1 = (float*)alloc((size_t)NB*NS*NAG*4);
    float* aggr2 = (float*)alloc((size_t)NB*NS*NAG*4);
    unsigned short* aggr1b = (unsigned short*)alloc((size_t)NB*NS*NAG*2);
    unsigned short* aggr2b = (unsigned short*)alloc((size_t)NB*NS*NAG*2);
    float* alpha = (float*)alloc((size_t)NS*NB*NB*4);
    float* rsum  = (float*)alloc((size_t)NS*NB*4);
    int*   amax  = (int*)alloc((size_t)NS*NB*4);
    float* rbuf  = (float*)alloc((size_t)NB*NS*NH*4);
    float* t1    = (float*)alloc((size_t)NB*2*NH*4);

    const size_t JOB = (size_t)NB * NS * N4H;
    auto cdiv = [](size_t a, size_t b) { return (unsigned)((a + b - 1) / b); };

    // weight / perspective bf16 conversions
    conv_bf16_k<<<cdiv((size_t)N4H*NDP,256),256,0,stream>>>(cWf, cWfb, N4H, ND, NDP);
    conv_bf16_k<<<cdiv((size_t)N4H*NDP,256),256,0,stream>>>(cWb, cWbb, N4H, ND, NDP);
    conv_bf16_k<<<cdiv((size_t)N4H*NH,256),256,0,stream>>>(cUf, cUfb, N4H, NH, NH);
    conv_bf16_k<<<cdiv((size_t)N4H*NH,256),256,0,stream>>>(cUb, cUbb, N4H, NH, NH);
    conv_bf16_k<<<cdiv((size_t)N4H*NAG,256),256,0,stream>>>(aWf, aWfb, N4H, NAG, NAG);
    conv_bf16_k<<<cdiv((size_t)N4H*NAG,256),256,0,stream>>>(aWb, aWbb, N4H, NAG, NAG);
    conv_bf16_k<<<cdiv((size_t)N4H*NH,256),256,0,stream>>>(aUf, aUfb, N4H, NH, NH);
    conv_bf16_k<<<cdiv((size_t)N4H*NH,256),256,0,stream>>>(aUb, aUbb, N4H, NH, NH);
    conv_bf16_k<<<cdiv((size_t)NL*NH,256),256,0,stream>>>(w3, w3b, NL, NH, NH);
    conv_bf16_k<<<cdiv((size_t)NL*NH,256),256,0,stream>>>(w4, w4b, NL, NH, NH);

    // embedding gather -> bf16 padded
    embed_bf16_k<<<cdiv((size_t)NB*NS*NDP,256),256,0,stream>>>(p_tok, emb, x1bf);
    embed_bf16_k<<<cdiv((size_t)NB*NS*NDP,256),256,0,stream>>>(h_tok, emb, x2bf);

    // context LSTM input projections (M=8192,N=512,K=320) x4 ; strips = 512*8 = 4096
    const unsigned gblk = cdiv((size_t)(NB*NS/16)*(N4H/64), 8);
    gemm_bf16_bias_k<<<gblk,256,0,stream>>>(x1bf, NDP, cWfb, NDP, cbf, Xc + 0*JOB, NB*NS, N4H, NDP);
    gemm_bf16_bias_k<<<gblk,256,0,stream>>>(x1bf, NDP, cWbb, NDP, cbb, Xc + 1*JOB, NB*NS, N4H, NDP);
    gemm_bf16_bias_k<<<gblk,256,0,stream>>>(x2bf, NDP, cWfb, NDP, cbf, Xc + 2*JOB, NB*NS, N4H, NDP);
    gemm_bf16_bias_k<<<gblk,256,0,stream>>>(x2bf, NDP, cWbb, NDP, cbb, Xc + 3*JOB, NB*NS, N4H, NDP);

    // context BiLSTM recurrence (both sequences, both directions in parallel)
    zero_state_k<<<cdiv(4*NB*NH,256),256,0,stream>>>(cst, hst, hprev);
    for (int t = 0; t < NS; ++t) {
        lstm_step_gemm_k<<<dim3(4,4),256,0,stream>>>(hprev, cUfb, cUbb, Xc, zbuf, t);
        lstm_step_gates_k<<<cdiv(4*NB*NH,256),256,0,stream>>>(zbuf, cst, hprev, hst,
            c1f, c1b, c2f, c2b, c1fb, c1bb, c2fb, c2bb, t, 1);
    }

    // norms
    bs_norm_k<<<cdiv(NB*NS,256),256,0,stream>>>(c1f, nb1f);
    bs_norm_k<<<cdiv(NB*NS,256),256,0,stream>>>(c1b, nb1b);
    bs_norm_k<<<cdiv(NB*NS,256),256,0,stream>>>(c2f, nb2f);
    bs_norm_k<<<cdiv(NB*NS,256),256,0,stream>>>(c2b, nb2b);
    w_norm_k<<<cdiv(NB*NL*NS,256),256,0,stream>>>(c1f, w3, wn1f3);
    w_norm_k<<<cdiv(NB*NL*NS,256),256,0,stream>>>(c2f, w3, wn2f3);
    w_norm_k<<<cdiv(NB*NL*NS,256),256,0,stream>>>(c1b, w4, wn1b4);
    w_norm_k<<<cdiv(NB*NL*NS,256),256,0,stream>>>(c2b, w4, wn2b4);

    // full match (cols 0..39): p2 = last-timestep hidden of the other sequence
    const unsigned mblk = cdiv(NB*NS*NL,256);
    full_match_k<<<mblk,256,0,stream>>>(c1f, c2f + (size_t)(NS-1)*NH, NS*NH, w1, aggr1, 0);
    full_match_k<<<mblk,256,0,stream>>>(c1b, c2b + (size_t)(NS-1)*NH, NS*NH, w2, aggr1, NL);
    full_match_k<<<mblk,256,0,stream>>>(c2f, c1f + (size_t)(NS-1)*NH, NS*NH, w1, aggr2, 0);
    full_match_k<<<mblk,256,0,stream>>>(c2b, c1b + (size_t)(NS-1)*NH, NS*NH, w2, aggr2, NL);

    // maxpool match (cols 40..79)
    const unsigned xblk = cdiv((size_t)NB*NL*(NS/16), 8);
    maxpool_match_k<<<xblk,256,0,stream>>>(c1fb, c2fb, w3b, wn1f3, wn2f3, aggr1, 2*NL);
    maxpool_match_k<<<xblk,256,0,stream>>>(c2fb, c1fb, w3b, wn2f3, wn1f3, aggr2, 2*NL);
    maxpool_match_k<<<xblk,256,0,stream>>>(c1bb, c2bb, w4b, wn1b4, wn2b4, aggr1, 3*NL);
    maxpool_match_k<<<xblk,256,0,stream>>>(c2bb, c1bb, w4b, wn2b4, wn1b4, aggr2, 3*NL);

    // attentive match (cols 80..159)
    auto attentive = [&](const float* p1f, const float* p2f,
                         const unsigned short* p1bf, const unsigned short* p2bf,
                         const float* n1, const float* n2,
                         const float* watt, const float* wmax, float* aggr,
                         int colA, int colM) {
        att_alpha_k<<<cdiv(NS*4,8),256,0,stream>>>(p1bf, p2bf, n1, n2, alpha);
        att_soft_k<<<cdiv(NS*NB,256),256,0,stream>>>(alpha, rsum, amax);
        att_h_k<<<cdiv((size_t)NB*NS*NH,256),256,0,stream>>>(alpha, rsum, p2f, rbuf);
        att_match_k<<<mblk,256,0,stream>>>(rbuf, p1f, watt, aggr, colA);
        att_max_k<<<mblk,256,0,stream>>>(amax, p2f, p1f, wmax, aggr, colM);
    };
    attentive(c1f, c2f, c1fb, c2fb, nb1f, nb2f, w5, w7, aggr1, 4*NL, 6*NL);
    attentive(c1b, c2b, c1bb, c2bb, nb1b, nb2b, w6, w8, aggr1, 5*NL, 7*NL);
    attentive(c2f, c1f, c2fb, c1fb, nb2f, nb1f, w5, w7, aggr2, 4*NL, 6*NL);
    attentive(c2b, c1b, c2bb, c1bb, nb2b, nb1b, w6, w8, aggr2, 5*NL, 7*NL);

    // aggregation BiLSTM
    conv_bf16_k<<<cdiv((size_t)NB*NS*NAG,256),256,0,stream>>>(aggr1, aggr1b, NB*NS, NAG, NAG);
    conv_bf16_k<<<cdiv((size_t)NB*NS*NAG,256),256,0,stream>>>(aggr2, aggr2b, NB*NS, NAG, NAG);
    gemm_bf16_bias_k<<<gblk,256,0,stream>>>(aggr1b, NAG, aWfb, NAG, abf, Xc + 0*JOB, NB*NS, N4H, NAG);
    gemm_bf16_bias_k<<<gblk,256,0,stream>>>(aggr1b, NAG, aWbb, NAG, abb, Xc + 1*JOB, NB*NS, N4H, NAG);
    gemm_bf16_bias_k<<<gblk,256,0,stream>>>(aggr2b, NAG, aWfb, NAG, abf, Xc + 2*JOB, NB*NS, N4H, NAG);
    gemm_bf16_bias_k<<<gblk,256,0,stream>>>(aggr2b, NAG, aWbb, NAG, abb, Xc + 3*JOB, NB*NS, N4H, NAG);

    zero_state_k<<<cdiv(4*NB*NH,256),256,0,stream>>>(cst, hst, hprev);
    for (int t = 0; t < NS; ++t) {
        lstm_step_gemm_k<<<dim3(4,4),256,0,stream>>>(hprev, aUfb, aUbb, Xc, zbuf, t);
        lstm_step_gates_k<<<cdiv(4*NB*NH,256),256,0,stream>>>(zbuf, cst, hprev, hst,
            c1f, c1b, c2f, c2b, c1fb, c1bb, c2fb, c2bb, t, 0);
    }

    // final FC: cat = [g1f g1b g2f g2b] = hst jobs 0..3
    fc1_k<<<cdiv(NB*2*NH,256),256,0,stream>>>(hst, f1W, f1b, t1);
    fc2_k<<<cdiv(NB*NC,256),256,0,stream>>>(t1, f2W, f2b, (float*)d_out);
}
